// ShortTermInterestExtractor_41772851921242
// MI455X (gfx1250) — matrix operand order repaired and checked by reading, not compiled, not run on hardware
//
#include <hip/hip_runtime.h>
#include <cmath>

typedef __attribute__((ext_vector_type(16))) _Float16     v16h;
typedef __attribute__((ext_vector_type(8)))  float        v8f;
typedef __attribute__((ext_vector_type(8)))  unsigned int v8u;
typedef __attribute__((ext_vector_type(8)))  _Float16     v8h;
typedef __attribute__((ext_vector_type(4)))  int          v4i;

#define HID 64
#define LDA 72   // halves per row in LDS A-tiles (padded; 144B row stride, 16B-aligned)
#define LDG 65   // floats per row in LDS gate buffer (padded)

#if defined(__gfx1250__) && __has_builtin(__builtin_amdgcn_global_load_async_to_lds_b128) && \
    __has_builtin(__builtin_amdgcn_s_wait_asynccnt)
#define HAVE_ASYNC_LDS 1
#else
#define HAVE_ASYNC_LDS 0
#endif

// Load a 16x32 f16 A-fragment from a row-major LDS tile (16 rows x LDA halves).
// ISA layout (16-bit A 16x32): lane L: M=L&15, halfgroup=L>>4;
// dword dw<4 -> K = hg*8 + dw*2 ; dw>=4 -> K = 16 + hg*8 + (dw-4)*2.
__device__ __forceinline__ v16h load_a_frag(const _Float16* tile, int lane, int kOff) {
  const int M = lane & 15, hg = lane >> 4;
  const unsigned int* t32 = (const unsigned int*)(tile + M * LDA); // 144B rows, 4B aligned
  v8u u;
#pragma unroll
  for (int dw = 0; dw < 8; ++dw) {
    const int k = (dw < 4) ? (hg * 8 + dw * 2) : (16 + hg * 8 + (dw - 4) * 2);
    u[dw] = t32[(kOff + k) >> 1];
  }
  return __builtin_bit_cast(v16h, u);
}

// Build a 32x16 f16 B-fragment (B = W^T slice) from row-major f32 W (192x64).
// B[k][n] = W[nBase+n][kOff+k]. ISA layout: lane L: N=L&15, hg=L>>4;
// dword dw<4 -> K = hg*8 + dw*2 ; dw>=4 -> K = 16 + hg*8 + (dw-4)*2.
__device__ __forceinline__ v16h load_b_frag(const float* W, int nBase, int kOff, int lane) {
  const int n = nBase + (lane & 15), hg = lane >> 4;
  const float* Wr = W + (size_t)n * HID + kOff + hg * 8;
  v16h f;
#pragma unroll
  for (int i = 0; i < 16; ++i) f[i] = (_Float16)Wr[(i < 8) ? i : (i + 8)];
  return f;
}

__device__ __forceinline__ float sigmoidf_(float x) { return 1.f / (1.f + expf(-x)); }

// Persistent recurrent kernel: one block = 16 batch rows, 4 waves.
//   wave0: r-gate  (cols   0..63),  fused gi+gh (K=128 via x-A and h-A)
//   wave1: z-gate  (cols  64..127), fused gi+gh
//   wave2: gin     (cols 128..191), x-A only
//   wave3: ghn     (cols 128..191), h-A only
template <bool AUGRU>
__global__ __launch_bounds__(128) void recurrent_kernel(
    const float* __restrict__ xF32,       // (B,H,64)   [GRU input]
    const _Float16* __restrict__ oF16,    // (H,B,64)   [AUGRU input]
    const int* __restrict__ lengths,      // (B,)
    const float* __restrict__ W_i,        // (192,64)
    const float* __restrict__ W_h,        // (192,64)
    const float* __restrict__ b_i,        // (192,)
    const float* __restrict__ b_h,        // (192,)
    const float* __restrict__ pos_query,  // (B,64)     [GRU]
    const float* __restrict__ att,        // (B,H)      [AUGRU]
    _Float16* __restrict__ outSeq,        // (H,B,64)   [GRU out]
    float* __restrict__ scores,           // (B,H)      [GRU out]
    float* __restrict__ hOut,             // (B,64)     [AUGRU out]
    int Btot, int Hlen) {
  __shared__ alignas(16) _Float16 xTile[16 * LDA];
  __shared__ alignas(16) _Float16 hTile[16 * LDA];
  __shared__ float hF32[16 * HID];
  __shared__ float gbuf[4 * 16 * LDG];  // [gate][m][n]
  __shared__ float pq[16 * HID];
  __shared__ float sPart[16 * 8];
  __shared__ float attRow[16];

  const int tid = threadIdx.x;
  const int lane = tid & 31;
  // Scalarize the wave id so wave-class branches become SALU branches
  // (EXEC stays all-1s through the WMMA regions, per ISA requirement).
  const int wid = __builtin_amdgcn_readfirstlane(tid >> 5);
  const int b0 = blockIdx.x * 16;

  const bool useX = (wid != 3), useH = (wid != 2);
  const int chunk = (wid < 2) ? wid : 2;

  // Weight fragments live in VGPRs for the whole 200-step recurrence.
  v16h Bi[4][2], Bh[4][2];
  float biasv[4];
#pragma unroll
  for (int nt = 0; nt < 4; ++nt) {
    const int nb = chunk * 64 + nt * 16;
#pragma unroll
    for (int kc = 0; kc < 2; ++kc) {
      if (useX) Bi[nt][kc] = load_b_frag(W_i, nb, kc * 32, lane);
      if (useH) Bh[nt][kc] = load_b_frag(W_h, nb, kc * 32, lane);
    }
    float bv = 0.f;
    const int ncol = nb + (lane & 15);
    if (useX) bv += b_i[ncol];
    if (useH) bv += b_h[ncol];
    biasv[nt] = bv;
  }

  const int m = tid >> 3, j0 = (tid & 7) * 8;  // combine-stage element mapping
  const int len_m = lengths[b0 + m];

#pragma unroll
  for (int jj = 0; jj < 8; ++jj) {
    hF32[m * HID + j0 + jj] = 0.f;
    hTile[m * LDA + j0 + jj] = (_Float16)0.f;
    if (!AUGRU) pq[m * HID + j0 + jj] = pos_query[(size_t)(b0 + m) * HID + j0 + jj];
  }
  __syncthreads();

  for (int t = 0; t < Hlen; ++t) {
    // ---- stage 1: stage input tile (f16) into LDS ----
    if (!AUGRU) {
      const float* src = xF32 + ((size_t)(b0 + m) * Hlen + t) * HID + j0;
#pragma unroll
      for (int jj = 0; jj < 8; ++jj) xTile[m * LDA + j0 + jj] = (_Float16)src[jj];
      if (t + 1 < Hlen) __builtin_prefetch(src + HID, 0, 3);
    } else {
      const _Float16* src = oF16 + ((size_t)t * Btot + b0 + m) * HID + j0;
#if HAVE_ASYNC_LDS
      // 128 threads x b128 = whole 16x64 f16 tile via the async-to-LDS path,
      // tracked by ASYNCcnt and drained before the barrier.
      __builtin_amdgcn_global_load_async_to_lds_b128(
          (__attribute__((address_space(1))) v4i*)(uintptr_t)src,
          (__attribute__((address_space(3))) v4i*)(unsigned int)(uintptr_t)&xTile[m * LDA + j0],
          0, 0);
      if (tid < 16) attRow[tid] = att[(size_t)(b0 + tid) * Hlen + t];
      if (t + 1 < Hlen) __builtin_prefetch(src + (size_t)Btot * HID, 0, 3);
      __builtin_amdgcn_s_wait_asynccnt(0);
#else
#pragma unroll
      for (int jj = 0; jj < 8; ++jj) xTile[m * LDA + j0 + jj] = src[jj];
      if (tid < 16) attRow[tid] = att[(size_t)(b0 + tid) * Hlen + t];
      if (t + 1 < Hlen) __builtin_prefetch(src + (size_t)Btot * HID, 0, 3);
#endif
    }
    __syncthreads();

    // ---- stage 2: WMMA gate GEMMs (scalar-uniform branches, EXEC all-1s) ----
    v16h xA0, xA1, hA0, hA1;
    if (useX) { xA0 = load_a_frag(xTile, lane, 0); xA1 = load_a_frag(xTile, lane, 32); }
    if (useH) { hA0 = load_a_frag(hTile, lane, 0); hA1 = load_a_frag(hTile, lane, 32); }
#pragma unroll
    for (int nt = 0; nt < 4; ++nt) {
      v8f acc = {};
      if (useX) {
        acc = __builtin_amdgcn_wmma_f32_16x16x32_f16(false, xA0, false, Bi[nt][0], (short)0, acc, false, false);
        acc = __builtin_amdgcn_wmma_f32_16x16x32_f16(false, xA1, false, Bi[nt][1], (short)0, acc, false, false);
      }
      if (useH) {
        acc = __builtin_amdgcn_wmma_f32_16x16x32_f16(false, hA0, false, Bh[nt][0], (short)0, acc, false, false);
        acc = __builtin_amdgcn_wmma_f32_16x16x32_f16(false, hA1, false, Bh[nt][1], (short)0, acc, false, false);
      }
      // C layout: lane L holds N=L&15, rows M=(L>>4)*8 + r
      const int col = nt * 16 + (lane & 15);
      const int rbase = (lane >> 4) * 8;
#pragma unroll
      for (int r = 0; r < 8; ++r)
        gbuf[(wid * 16 + rbase + r) * LDG + col] = acc[r] + biasv[nt];
    }
    __syncthreads();

    // ---- stage 3: gates, state update, outputs ----
    const bool valid = (t < len_m);
    float a_t = 0.f;
    if (AUGRU) a_t = attRow[m];
    float dot = 0.f;
    v8h outv;
#pragma unroll
    for (int jj = 0; jj < 8; ++jj) {
      const int j = j0 + jj;
      const float gr = gbuf[(0 * 16 + m) * LDG + j];
      const float gz = gbuf[(1 * 16 + m) * LDG + j];
      const float gin = gbuf[(2 * 16 + m) * LDG + j];
      const float ghn = gbuf[(3 * 16 + m) * LDG + j];
      const float rg = sigmoidf_(gr);
      const float zg = sigmoidf_(gz);
      const float ng = tanhf(gin + rg * ghn);
      const float h = hF32[m * HID + j];
      float hn;
      if (!AUGRU) {
        hn = (1.f - zg) * ng + zg * h;
      } else {
        const float u = a_t * zg;
        hn = (1.f - u) * h + u * ng;
      }
      const float hnew = valid ? hn : h;
      hF32[m * HID + j] = hnew;
      hTile[m * LDA + j] = (_Float16)hnew;
      if (!AUGRU) {
        const _Float16 o = valid ? (_Float16)hn : (_Float16)0.f;
        outv[jj] = o;
        dot += (float)o * pq[m * HID + j];
      }
    }
    if (!AUGRU) {
      *(v8h*)(outSeq + ((size_t)t * Btot + b0 + m) * HID + j0) = outv;  // 16B aligned
      sPart[m * 8 + (tid & 7)] = dot;
      __syncthreads();
      if (tid < 16) {
        float s = 0.f;
#pragma unroll
        for (int p = 0; p < 8; ++p) s += sPart[tid * 8 + p];
        scores[(size_t)(b0 + tid) * Hlen + t] = s;
      }
    }
    __syncthreads();
  }

  if (AUGRU) {
#pragma unroll
    for (int jj = 0; jj < 8; ++jj)
      hOut[(size_t)(b0 + m) * HID + j0 + jj] = hF32[m * HID + j0 + jj];
  }
}

// One block per batch row; softmax over Hlen (<=256) scores. Padded steps have
// score 0 and participate, matching the reference.
__global__ __launch_bounds__(256) void softmax_kernel(const float* __restrict__ scores,
                                                      float* __restrict__ att,
                                                      int Btot, int Hlen) {
  __shared__ float red[256];
  const int b = blockIdx.x, tid = threadIdx.x;
  const float v = (tid < Hlen) ? scores[(size_t)b * Hlen + tid] : -INFINITY;
  red[tid] = v;
  __syncthreads();
  for (int s = 128; s > 0; s >>= 1) {
    if (tid < s) red[tid] = fmaxf(red[tid], red[tid + s]);
    __syncthreads();
  }
  const float mx = red[0];
  __syncthreads();
  const float e = (tid < Hlen) ? expf(v - mx) : 0.f;
  red[tid] = e;
  __syncthreads();
  for (int s = 128; s > 0; s >>= 1) {
    if (tid < s) red[tid] += red[tid + s];
    __syncthreads();
  }
  const float inv = 1.f / red[0];
  if (tid < Hlen) att[(size_t)b * Hlen + tid] = e * inv;
}

extern "C" void kernel_launch(void* const* d_in, const int* in_sizes, int n_in,
                              void* d_out, int out_size, void* d_ws, size_t ws_size,
                              hipStream_t stream) {
  const float* x     = (const float*)d_in[0];
  const float* pos   = (const float*)d_in[1];
  const int*   len   = (const int*)d_in[2];
  const float* W_ih  = (const float*)d_in[3];
  const float* W_hh  = (const float*)d_in[4];
  const float* b_ih  = (const float*)d_in[5];
  const float* b_hh  = (const float*)d_in[6];
  const float* aW_ih = (const float*)d_in[7];
  const float* aW_hh = (const float*)d_in[8];
  const float* ab_ih = (const float*)d_in[9];
  const float* ab_hh = (const float*)d_in[10];

  const int B = in_sizes[2];               // lengths: (B,)
  const int H = in_sizes[0] / (B * HID);   // x: (B,H,64)

  // Workspace layout: out (H,B,64) f16 | scores (B,H) f32 | att (B,H) f32
  char* ws = (char*)d_ws;
  _Float16* outSeq = (_Float16*)ws;
  size_t off = (size_t)H * B * HID * sizeof(_Float16);
  off = (off + 255) & ~(size_t)255;
  float* scores = (float*)(ws + off);
  off += (size_t)B * H * sizeof(float);
  off = (off + 255) & ~(size_t)255;
  float* att = (float*)(ws + off);

  dim3 grid(B / 16), block(128);
  recurrent_kernel<false><<<grid, block, 0, stream>>>(
      x, nullptr, len, W_ih, W_hh, b_ih, b_hh, pos, nullptr,
      outSeq, scores, nullptr, B, H);
  softmax_kernel<<<dim3(B), dim3(256), 0, stream>>>(scores, att, B, H);
  recurrent_kernel<true><<<grid, block, 0, stream>>>(
      nullptr, outSeq, len, aW_ih, aW_hh, ab_ih, ab_hh, nullptr, att,
      nullptr, nullptr, (float*)d_out, B, H);
}
